// CausalAttention_44478681317653
// MI455X (gfx1250) — matrix operand order repaired and checked by reading, not compiled, not run on hardware
//
#include <hip/hip_runtime.h>

// Problem constants from the reference: B=4, S=4096, D=512
#define Bb 4
#define Ss 4096
#define Dd 512
#define NROW (Bb * Ss)                     // 16384 flattened rows
#define NELEM ((size_t)NROW * (size_t)Dd)  // 8,388,608 elements per tensor

typedef __attribute__((ext_vector_type(16))) __bf16 v16bf;
typedef __attribute__((ext_vector_type(8)))  float  v8f;

union Frag {
    uint4          q[2];
    __bf16         h[16];
    unsigned short us[16];
    v16bf          v;
};

__device__ __forceinline__ unsigned short bf2us(__bf16 h) {
    union { __bf16 b; unsigned short u; } x;
    x.b = h;
    return x.u;
}

// 16 contiguous bf16 values (32B) -> one fragment register block
__device__ __forceinline__ v16bf load_bf16_contig(const unsigned short* p) {
    Frag f;
    f.q[0] = *(const uint4*)(p);
    f.q[1] = *(const uint4*)(p + 8);
    return f.v;
}

__device__ __forceinline__ v8f wmma_bf16(v16bf a, v16bf b, v8f c) {
    // D = A(16x32 bf16) * B(32x16 bf16) + C(16x16 f32)
    return __builtin_amdgcn_wmma_f32_16x16x32_bf16(
        /*neg_a=*/false, a, /*neg_b=*/false, b,
        /*c_mod=*/(short)0, c, /*reuse_a=*/false, /*reuse_b=*/false);
}

// Use native f32->bf16 conversion (v_cvt_*bf16_f32, RNE) instead of manual bit ops
__device__ __forceinline__ void cvt4(float4 f, __bf16* d) {
    d[0] = (__bf16)f.x; d[1] = (__bf16)f.y;
    d[2] = (__bf16)f.z; d[3] = (__bf16)f.w;
}

// ---------------------------------------------------------------------------
// Kernel 1: fused QKV projection.  out = x @ W^T + b, converted to bf16.
//   which = blockIdx.z : 0 -> Q (pre-scaled by 1/sqrt(D)), 1 -> K, 2 -> V^T.
// Each wave computes one 16x16 tile; 4 waves per block cover N-quad of 64.
// ---------------------------------------------------------------------------
__global__ void __launch_bounds__(128, 2)
qkv_proj_kernel(const float* __restrict__ x,
                const float* __restrict__ Wq, const float* __restrict__ bq,
                const float* __restrict__ Wk, const float* __restrict__ bk,
                const float* __restrict__ Wv, const float* __restrict__ bv,
                unsigned short* __restrict__ Qbf,
                unsigned short* __restrict__ Kbf,
                unsigned short* __restrict__ Vt) {
    const int lane = threadIdx.x & 31;
    const int w    = threadIdx.x >> 5;   // wave 0..3
    const int hi   = lane >> 4;          // lane half
    const int ln   = lane & 15;

    const int m0    = blockIdx.x * 16;
    const int n0    = blockIdx.y * 64 + w * 16;
    const int which = blockIdx.z;

    const float* W    = (which == 0) ? Wq : (which == 1) ? Wk : Wv;
    const float* bias = (which == 0) ? bq : (which == 1) ? bk : bv;

    const float* xrow = x + (size_t)(m0 + ln) * Dd;  // A row (M = ln)
    const float* wrow = W + (size_t)(n0 + ln) * Dd;  // B column n = ln is W row e

    v8f c;
#pragma unroll
    for (int r = 0; r < 8; ++r) c[r] = 0.0f;

    for (int k0 = 0; k0 < Dd; k0 += 32) {
        if (k0 + 32 < Dd) {
            __builtin_prefetch(xrow + k0 + 32 + hi * 8, 0, 3);
            __builtin_prefetch(wrow + k0 + 32 + hi * 16, 0, 3);
        }
        // A fragment: lane half 0 holds K 0..15, half 1 holds K 16..31 (contiguous)
        Frag a;
        {
            const float* p = xrow + k0 + hi * 16;
            float4 f0 = *(const float4*)(p);
            float4 f1 = *(const float4*)(p + 4);
            float4 f2 = *(const float4*)(p + 8);
            float4 f3 = *(const float4*)(p + 12);
            cvt4(f0, a.h + 0);  cvt4(f1, a.h + 4);
            cvt4(f2, a.h + 8);  cvt4(f3, a.h + 12);
        }
        // B fragment: lane half 0 holds K 0..15, half 1 holds K 16..31 (contiguous)
        Frag bfr;
        {
            const float* p = wrow + k0 + hi * 16;
            float4 f0 = *(const float4*)(p);
            float4 f1 = *(const float4*)(p + 4);
            float4 f2 = *(const float4*)(p + 8);
            float4 f3 = *(const float4*)(p + 12);
            cvt4(f0, bfr.h + 0);  cvt4(f1, bfr.h + 4);
            cvt4(f2, bfr.h + 8);  cvt4(f3, bfr.h + 12);
        }
        c = wmma_bf16(a.v, bfr.v, c);
    }

    const int   col    = n0 + ln;
    const float bval   = bias[col];
    const float qscale = (which == 0) ? 0.04419417382415922f : 1.0f;  // 1/sqrt(512)

#pragma unroll
    for (int r = 0; r < 8; ++r) {
        const int   row = m0 + r + 8 * hi;   // C layout: M = r (+8 for upper lanes)
        const float v   = (c[r] + bval) * qscale;
        const unsigned short hbits = bf2us((__bf16)v);
        if (which == 0) {
            Qbf[(size_t)row * Dd + col] = hbits;
        } else if (which == 1) {
            Kbf[(size_t)row * Dd + col] = hbits;
        } else {
            // store V transposed: Vt[b][d][s] so P*V B-fragments are contiguous
            const int bb = row >> 12;            // row / S
            const int s  = row & (Ss - 1);       // row % S
            Vt[((size_t)(bb * Dd + col)) * Ss + s] = hbits;
        }
    }
}

// ---------------------------------------------------------------------------
// Kernel 2: flash attention over 16 query rows per block, 8 waves.
// Wave w owns d-slice [64w, 64w+64). Per 32-key step:
//   partial QK^T per wave -> LDS reduction spread over all 256 threads
//   (deterministic fixed-order sums) -> online softmax on 32 lanes of wave 0
//   -> P (bf16, row-major in LDS, read back as A fragments) -> P*V WMMAs.
// ---------------------------------------------------------------------------
__global__ void __launch_bounds__(256, 1)
attn_kernel(const unsigned short* __restrict__ Qbf,
            const unsigned short* __restrict__ Kbf,
            const unsigned short* __restrict__ Vt,
            const unsigned char* __restrict__ mask,
            float* __restrict__ out) {
    __shared__ float          sPart[8][16][32];   // per-wave partial scores (16 KB)
    __shared__ float          sScore[16][32];     // reduced scores (2 KB)
    __shared__ unsigned short sP[16][32];         // P tile, bf16 row-major (1 KB)
    __shared__ float          sCorr[16];
    __shared__ float          sL[16];

    const int tid  = threadIdx.x;
    const int lane = tid & 31;
    const int w    = tid >> 5;       // wave 0..7
    const int hi   = lane >> 4;
    const int ln   = lane & 15;

    const int R0 = blockIdx.x * 16;  // flattened query-row base
    const int b  = R0 >> 12;         // batch
    const int dw = w * 64;           // this wave's d-slice base
    const int kvBase = b * Ss;       // kv row base in flattened [B*S, D]

    const float NEG_INF = -__builtin_inff();

    // Persistent Q fragments for our d-slice (2 K-chunks of 32)
    v16bf aQ[2];
#pragma unroll
    for (int kc = 0; kc < 2; ++kc) {
        const unsigned short* qp =
            Qbf + (size_t)(R0 + ln) * Dd + dw + kc * 32 + hi * 16;
        Frag f;
        f.q[0] = *(const uint4*)(qp);
        f.q[1] = *(const uint4*)(qp + 8);
        aQ[kc] = f.v;
    }

    v8f O[4];
#pragma unroll
    for (int nt = 0; nt < 4; ++nt)
#pragma unroll
        for (int r = 0; r < 8; ++r) O[nt][r] = 0.0f;

    float mOld = NEG_INF;   // live in lanes 0..31 of wave 0 (halves replicated)
    float lRow = 0.0f;

    // Base pointers for this wave's K / Vt streams
    const unsigned short* kBase = Kbf + (size_t)(kvBase + ln) * Dd + dw + hi * 16;
    const unsigned short* vBase = Vt + ((size_t)(b * Dd + dw + ln)) * Ss + hi * 16;

    for (int j0 = 0; j0 < Ss; j0 += 32) {
        // Prefetch next kv tile into cache (L2-resident stream -> hides L0 miss)
        if (j0 + 32 < Ss) {
            __builtin_prefetch(kBase + (size_t)(j0 + 32) * Dd, 0, 3);
            __builtin_prefetch(vBase + (j0 + 32), 0, 3);
        }

        // ---- partial scores over this wave's 64-wide d slice ----
        v8f C1, C2;
#pragma unroll
        for (int r = 0; r < 8; ++r) { C1[r] = 0.0f; C2[r] = 0.0f; }

#pragma unroll
        for (int kc = 0; kc < 2; ++kc) {
            const unsigned short* kp1 = kBase + (size_t)j0 * Dd + kc * 32;
            const unsigned short* kp2 = kp1 + (size_t)16 * Dd;
            v16bf B1 = load_bf16_contig(kp1);   // keys j0..j0+15
            v16bf B2 = load_bf16_contig(kp2);   // keys j0+16..j0+31
            C1 = wmma_bf16(aQ[kc], B1, C1);
            C2 = wmma_bf16(aQ[kc], B2, C2);
        }
#pragma unroll
        for (int r = 0; r < 8; ++r) {
            sPart[w][r + 8 * hi][ln]      = C1[r];
            sPart[w][r + 8 * hi][16 + ln] = C2[r];
        }
        __syncthreads();

        // ---- 8-way reduction spread across all 256 threads (fixed order) ----
        {
            const int row = tid >> 4;            // 2 entries per thread
            const int col = (tid & 15) * 2;
            float2 acc = *(const float2*)&sPart[0][row][col];
#pragma unroll
            for (int ww = 1; ww < 8; ++ww) {
                float2 t = *(const float2*)&sPart[ww][row][col];
                acc.x += t.x;
                acc.y += t.y;
            }
            *(float2*)&sScore[row][col] = acc;
        }
        __syncthreads();

        // ---- online softmax on wave 0: lane = row (0..15) x column-half ----
        if (tid < 32) {
            const int r = lane & 15;
            const int h = lane >> 4;
            float s[16];
            {
                float4 s0 = *(const float4*)&sScore[r][h * 16 + 0];
                float4 s1 = *(const float4*)&sScore[r][h * 16 + 4];
                float4 s2 = *(const float4*)&sScore[r][h * 16 + 8];
                float4 s3 = *(const float4*)&sScore[r][h * 16 + 12];
                s[0] = s0.x;  s[1] = s0.y;  s[2] = s0.z;  s[3] = s0.w;
                s[4] = s1.x;  s[5] = s1.y;  s[6] = s1.z;  s[7] = s1.w;
                s[8] = s2.x;  s[9] = s2.y;  s[10] = s2.z; s[11] = s2.w;
                s[12] = s3.x; s[13] = s3.y; s[14] = s3.z; s[15] = s3.w;
            }
            const unsigned char* mrow = mask + b * Ss + j0 + h * 16;
#pragma unroll
            for (int i = 0; i < 16; ++i)
                if (mrow[i]) s[i] = NEG_INF;

            float mTile = s[0];
#pragma unroll
            for (int i = 1; i < 16; ++i) mTile = fmaxf(mTile, s[i]);
            mTile = fmaxf(mTile, __shfl_xor(mTile, 16, 32));  // combine halves
            const float mNew = fmaxf(mOld, mTile);
            const float corr = (mNew == NEG_INF) ? 1.0f : __expf(mOld - mNew);

            float sum = 0.0f;
            Frag pf;
#pragma unroll
            for (int i = 0; i < 16; ++i) {
                const float p = (s[i] == NEG_INF) ? 0.0f : __expf(s[i] - mNew);
                sum += p;
                pf.h[i] = (__bf16)p;
            }
            sum += __shfl_xor(sum, 16, 32);                   // full-row sum
            lRow = lRow * corr + sum;
            mOld = mNew;

            *(uint4*)&sP[r][h * 16 + 0] = pf.q[0];
            *(uint4*)&sP[r][h * 16 + 8] = pf.q[1];
            if (h == 0) sCorr[r] = corr;
        }
        __syncthreads();

        // ---- rescale O and accumulate P * V over our d slice ----
        float cr[8];
#pragma unroll
        for (int r = 0; r < 8; ++r) cr[r] = sCorr[r + 8 * hi];

        Frag pf;  // P as A fragment: row m = ln, K packing by lane half
        pf.q[0] = *(const uint4*)&sP[ln][hi * 16];
        pf.q[1] = *(const uint4*)&sP[ln][hi * 16 + 8];
        const v16bf P = pf.v;

#pragma unroll
        for (int nt = 0; nt < 4; ++nt) {
#pragma unroll
            for (int r = 0; r < 8; ++r) O[nt][r] *= cr[r];
            // B fragment: column d = dw+nt*16+ln, keys contiguous thanks to Vt
            v16bf Vf = load_bf16_contig(vBase + (size_t)(nt * 16) * Ss + j0);
            O[nt] = wmma_bf16(P, Vf, O[nt]);
        }
    }

    if (tid < 16) sL[tid] = lRow;
    __syncthreads();

    float il[8];
#pragma unroll
    for (int r = 0; r < 8; ++r) il[r] = 1.0f / sL[r + 8 * hi];

#pragma unroll
    for (int nt = 0; nt < 4; ++nt) {
#pragma unroll
        for (int r = 0; r < 8; ++r) {
            out[(size_t)(R0 + r + 8 * hi) * Dd + dw + nt * 16 + ln] =
                O[nt][r] * il[r];
        }
    }
}

// ---------------------------------------------------------------------------
extern "C" void kernel_launch(void* const* d_in, const int* in_sizes, int n_in,
                              void* d_out, int out_size, void* d_ws, size_t ws_size,
                              hipStream_t stream) {
    const float*         x    = (const float*)d_in[0];
    const unsigned char* mask = (const unsigned char*)d_in[1];
    const float*         Wq   = (const float*)d_in[2];
    const float*         bq   = (const float*)d_in[3];
    const float*         Wk   = (const float*)d_in[4];
    const float*         bk   = (const float*)d_in[5];
    const float*         Wv   = (const float*)d_in[6];
    const float*         bv   = (const float*)d_in[7];
    float*               y    = (float*)d_out;

    // Workspace: Q (bf16, pre-scaled), K (bf16), V^T (bf16) — 50.3 MB total
    unsigned short* Qbf = (unsigned short*)d_ws;
    unsigned short* Kbf = Qbf + NELEM;
    unsigned short* Vt  = Kbf + NELEM;

    (void)in_sizes; (void)n_in; (void)out_size; (void)ws_size;

    dim3 gProj(NROW / 16, Dd / 64, 3);   // (1024, 8, 3)
    qkv_proj_kernel<<<gProj, 128, 0, stream>>>(x, Wq, bq, Wk, bk, Wv, bv,
                                               Qbf, Kbf, Vt);

    dim3 gAttn(NROW / 16);               // 1024 blocks, 16 query rows each
    attn_kernel<<<gAttn, 256, 0, stream>>>(Qbf, Kbf, Vt, mask, y);
}
